// GraphormerFishAttention_74844100100342
// MI455X (gfx1250) — compile-verified, matrix operand order checked by reading, not run on hardware
//
#include <hip/hip_runtime.h>
#include <hip/hip_fp16.h>

// ---------------------------------------------------------------------------
// GraphormerFishAttention for MI455X (gfx1250, wave32, WMMA f16->f32)
// B=16, N=512, H=512, G=8, L=16, D=64
// ---------------------------------------------------------------------------

typedef __attribute__((ext_vector_type(16))) _Float16 v16h;
typedef __attribute__((ext_vector_type(8)))  _Float16 v8h;
typedef __attribute__((ext_vector_type(8)))  float    v8f;

#define CB 16
#define CN 512
#define CH 512
#define CG 8
#define CL 16
#define CD 64
#define CM_TOT (CB * CN)          // 8192 rows of x
#define SCALE_F 0.04419417382415922f

__device__ __forceinline__ v16h cat16(v8h lo, v8h hi) {
    v16h r;
#pragma unroll
    for (int i = 0; i < 8; ++i) { r[i] = lo[i]; r[i + 8] = hi[i]; }
    return r;
}

__device__ __forceinline__ v8f wmma_f16(v16h a, v16h b, v8f c) {
    // D = A(16x32 f16) x B(32x16 f16) + C(16x16 f32)
    return __builtin_amdgcn_wmma_f32_16x16x32_f16(false, a, false, b,
                                                  (short)0, c, false, false);
}

__device__ __forceinline__ float mishf(float x) {
    // mish(x) = x * tanh(softplus(x)) = x * ((1+e^x)^2 - 1)/((1+e^x)^2 + 1)
    float ex = __expf(x);
    float p  = 1.0f + ex;
    float p2 = p * p;
    float t  = (p2 - 1.0f) / (p2 + 1.0f);
    return (x > 15.0f) ? x : x * t;
}

// ---------------------------------------------------------------------------
// Conversion kernels
// ---------------------------------------------------------------------------
__global__ void cvt_copy_f16(const float* __restrict__ in,
                             _Float16* __restrict__ out, int n) {
    int i = blockIdx.x * 256 + threadIdx.x;
    if (i < n) out[i] = (_Float16)in[i];
}

// in: rows(K) x cols(N) row-major  ->  out: cols x rows row-major (transposed)
__global__ void cvt_transpose_f16(const float* __restrict__ in,
                                  _Float16* __restrict__ out,
                                  int rows, int cols) {
    int i = blockIdx.x * 256 + threadIdx.x;
    if (i >= rows * cols) return;
    int c = i / rows;
    int r = i - c * rows;
    out[i] = (_Float16)in[r * cols + c];
}

// ---------------------------------------------------------------------------
// Kernel 1: fused q/k/v projection GEMM.
//   A = x_f16 (8192 x 512), virtual B = [WqT | WkT | WvT] -> 2048 columns.
//   q16,k16 row-major; v stored pre-transposed vT[b][l][d][m] (f16).
// Grid: (M/16, 2048/16/8), 256 threads (8 waves), one 16x16 tile per wave.
// ---------------------------------------------------------------------------
__global__ __launch_bounds__(256) void qkv_gemm(
    const _Float16* __restrict__ x16,
    const _Float16* __restrict__ wqT,   // (512 x 512)  [col][k]
    const _Float16* __restrict__ wkT,   // (512 x 512)
    const _Float16* __restrict__ wvT,   // (1024 x 512)
    const float*    __restrict__ bv,    // (1024)
    _Float16* __restrict__ q16,         // (8192 x 512)
    _Float16* __restrict__ k16,         // (8192 x 512)
    _Float16* __restrict__ vT16)        // (16 x 16 x 64 x 512)
{
    __shared__ __align__(16) _Float16 lds_a[16 * CH];   // 16KB A tile
    const int t    = threadIdx.x;
    const int row0 = blockIdx.x * 16;

    // stage A tile (16 x 512 halves)
    {
        const _Float16* src = x16 + (size_t)row0 * CH;
#pragma unroll
        for (int i = 0; i < 4; ++i) {
            int idx = (t + i * 256) * 8;
            *(v8h*)&lds_a[idx] = *(const v8h*)&src[idx];
        }
    }
    __syncthreads();

    const int wave  = t >> 5;
    const int lane  = t & 31;
    const int lhalf = lane & 15;
    const int koff  = (lane < 16) ? 0 : 8;
    const int ctile = blockIdx.y * 8 + wave;      // 0..127
    const int col0  = ctile * 16;

    const _Float16* wT; int cBase;
    if (col0 < 512)        { wT = wqT; cBase = col0; }
    else if (col0 < 1024)  { wT = wkT; cBase = col0 - 512; }
    else                   { wT = wvT; cBase = col0 - 1024; }
    const _Float16* brow = wT + (size_t)(cBase + lhalf) * CH;

    v8f acc = {0.f, 0.f, 0.f, 0.f, 0.f, 0.f, 0.f, 0.f};
#pragma unroll 4
    for (int k0 = 0; k0 < CH; k0 += 32) {
        v8h alo = *(const v8h*)&lds_a[lhalf * CH + k0 + koff];
        v8h ahi = *(const v8h*)&lds_a[lhalf * CH + k0 + koff + 16];
        v8h blo = *(const v8h*)&brow[k0 + koff];
        v8h bhi = *(const v8h*)&brow[k0 + koff + 16];
        acc = wmma_f16(cat16(alo, ahi), cat16(blo, bhi), acc);
    }

    const int mAdd = (lane < 16) ? 0 : 8;
    if (col0 < 1024) {
        _Float16* dst = (col0 < 512) ? q16 : k16;
        int cc = cBase + lhalf;
#pragma unroll
        for (int r = 0; r < 8; ++r) {
            int row = row0 + r + mAdd;
            dst[(size_t)row * CH + cc] = (_Float16)acc[r];
        }
    } else {
        int c = cBase + lhalf;                    // 0..1023 into L*D
        float bias = bv[c];
        int l = c >> 6, d = c & 63;
#pragma unroll
        for (int r = 0; r < 8; ++r) {
            int row = row0 + r + mAdd;
            int b = row >> 9, m = row & 511;
            vT16[(size_t)(((b * CL + l) * CD + d)) * CN + m] =
                (_Float16)(acc[r] + bias);
        }
    }
}

// ---------------------------------------------------------------------------
// Kernel 2: fused attention.
//   scores (WMMA) -> +sigma^2*eps -> G->L MLP with mish -> *SCALE + prior
//   -> softmax over L -> o accumulation (WMMA).
// Grid: B*N/16 blocks; 512 threads = 16 waves.
// Phase A: wave = (mh, g)  computes 16n x 16m score tile for head g.
// Phase C: one thread per (n, m) element (16 x 32 = 512 elements).
// Phase D: wave = l accumulates o[l] (16 x 64) with K=32 WMMAs.
// ---------------------------------------------------------------------------
#define OFF_SIG2 0
#define OFF_WP1  8
#define OFF_BP1  136
#define OFF_WP2  152
#define OFF_BP2  408
#define NCONST   424

__global__ __launch_bounds__(512) void attn_fused(
    const _Float16* __restrict__ q16,   // (B*N x 512)
    const _Float16* __restrict__ k16,   // (B*N x 512)
    const _Float16* __restrict__ vT16,  // (B x L x D x N)
    const float* __restrict__ eps,      // (B x N x N x G)
    const float* __restrict__ prior,    // (B x L x N x N)
    const float* __restrict__ sigma,    // (G)
    const float* __restrict__ Wp1,      // (G x L)
    const float* __restrict__ bp1,      // (L)
    const float* __restrict__ Wp2,      // (L x L)
    const float* __restrict__ bp2,      // (L)
    _Float16* __restrict__ o16)         // (B*N x 1024)
{
    __shared__ __align__(16) _Float16 lds_q[16 * CH];          // 16KB
    __shared__ __align__(16) float    lds_gk[2 * CG * 256];    // 16KB [mh][g][M*16+N]
    __shared__ __align__(16) _Float16 lds_at[CL * 16 * 32];    // 16KB [l][n][mm]
    __shared__            float    lds_c[NCONST];

    const int t    = threadIdx.x;
    const int wave = t >> 5;
    const int lane = t & 31;
    const int bb   = blockIdx.x >> 5;
    const int n0   = (blockIdx.x & 31) * 16;

    // small constants -> LDS
    if (t < 8)        lds_c[OFF_SIG2 + t] = sigma[t] * sigma[t];
    else if (t < 136) lds_c[t] = Wp1[t - OFF_WP1];
    else if (t < 152) lds_c[t] = bp1[t - OFF_BP1];
    else if (t < 408) lds_c[t] = Wp2[t - OFF_WP2];
    else if (t < 424) lds_c[t] = bp2[t - OFF_BP2];

    // q tile (16 x 512 halves) -> LDS
    {
        const _Float16* qsrc = q16 + (size_t)(bb * CN + n0) * CH;
        *(v8h*)&lds_q[t * 16]     = *(const v8h*)&qsrc[t * 16];
        *(v8h*)&lds_q[t * 16 + 8] = *(const v8h*)&qsrc[t * 16 + 8];
    }
    __syncthreads();

    const int g     = wave & 7;
    const int mh    = wave >> 3;
    const int l     = wave;               // phase D role
    const int lhalf = lane & 15;
    const int koff  = (lane < 16) ? 0 : 8;
    const int mAdd  = (lane < 16) ? 0 : 8;
    const int el_n  = t >> 5;             // phase C role
    const int el_m  = t & 31;

    v8f o_acc[4];
#pragma unroll
    for (int i = 0; i < 4; ++i)
        o_acc[i] = (v8f){0.f, 0.f, 0.f, 0.f, 0.f, 0.f, 0.f, 0.f};

    for (int m0 = 0; m0 < CN; m0 += 32) {
        // ---------------- phase A: score tile (head g, m-half mh) ----------
        {
            v8f s = {0.f, 0.f, 0.f, 0.f, 0.f, 0.f, 0.f, 0.f};
            const _Float16* krow =
                k16 + (size_t)(bb * CN + m0 + mh * 16 + lhalf) * CH + g * CD;
#pragma unroll
            for (int kk = 0; kk < CD; kk += 32) {
                v8h alo = *(const v8h*)&lds_q[lhalf * CH + g * CD + kk + koff];
                v8h ahi = *(const v8h*)&lds_q[lhalf * CH + g * CD + kk + koff + 16];
                v8h blo = *(const v8h*)&krow[kk + koff];
                v8h bhi = *(const v8h*)&krow[kk + koff + 16];
                s = wmma_f16(cat16(alo, ahi), cat16(blo, bhi), s);
            }
            int base = (mh * CG + g) * 256 + mAdd * 16 + lhalf;
#pragma unroll
            for (int r = 0; r < 8; ++r) lds_gk[base + r * 16] = s[r];
        }
        __syncthreads();

        // ---------------- phase C: per-element noise + MLP + softmax -------
        {
            float sg[CG];
            bool pad = true;
            int gbase = (el_m >> 4) * (CG * 256) + el_n * 16 + (el_m & 15);
#pragma unroll
            for (int gg = 0; gg < CG; ++gg) {
                float v = lds_gk[gbase + gg * 256];
                pad = pad && (v == 0.0f);
                sg[gg] = v;
            }
            const float* ep =
                eps + (size_t)(((bb * CN + n0 + el_n) * CN) + m0 + el_m) * CG;
#pragma unroll
            for (int gg = 0; gg < CG; ++gg)
                sg[gg] = fmaf(lds_c[OFF_SIG2 + gg], ep[gg], sg[gg]);

            float h[CL];
#pragma unroll
            for (int j = 0; j < CL; ++j) {
                float a = lds_c[OFF_BP1 + j];
#pragma unroll
                for (int gg = 0; gg < CG; ++gg)
                    a = fmaf(sg[gg], lds_c[OFF_WP1 + gg * CL + j], a);
                h[j] = mishf(a);
            }
            float av[CL];
#pragma unroll
            for (int ll = 0; ll < CL; ++ll) {
                float a = lds_c[OFF_BP2 + ll];
#pragma unroll
                for (int j = 0; j < CL; ++j)
                    a = fmaf(h[j], lds_c[OFF_WP2 + j * CL + ll], a);
                av[ll] = a;
            }
            // *SCALE + prior[b, l, n, m]  (coalesced over m per l)
            const float* pr = prior +
                (size_t)(bb * CL) * (CN * CN) + (size_t)(n0 + el_n) * CN + m0 + el_m;
#pragma unroll
            for (int ll = 0; ll < CL; ++ll)
                av[ll] = fmaf(av[ll], SCALE_F, pr[(size_t)ll * (CN * CN)]);
            // softmax over the 16 local heads
            float mx = av[0];
#pragma unroll
            for (int ll = 1; ll < CL; ++ll) mx = fmaxf(mx, av[ll]);
            float ssum = 0.f;
#pragma unroll
            for (int ll = 0; ll < CL; ++ll) {
                av[ll] = __expf(av[ll] - mx);
                ssum += av[ll];
            }
            float inv = 1.0f / ssum;
#pragma unroll
            for (int ll = 0; ll < CL; ++ll) {
                float o = pad ? 0.0f : av[ll] * inv;   // pad -> -inf -> nan_to_num(0)
                lds_at[(ll * 16 + el_n) * 32 + el_m] = (_Float16)o;
            }
        }
        __syncthreads();

        // ---------------- phase D: o[l] += a_l(16x32) @ v_l(32x64) ---------
        {
            v8h alo = *(const v8h*)&lds_at[(l * 16 + lhalf) * 32 + koff];
            v8h ahi = *(const v8h*)&lds_at[(l * 16 + lhalf) * 32 + koff + 16];
            v16h Am = cat16(alo, ahi);
            const _Float16* vbase =
                vT16 + (size_t)((bb * CL + l) * CD) * CN + m0;
#pragma unroll
            for (int dt = 0; dt < 4; ++dt) {
                const _Float16* vr = vbase + (size_t)(dt * 16 + lhalf) * CN;
                v8h blo = *(const v8h*)&vr[koff];
                v8h bhi = *(const v8h*)&vr[koff + 16];
                o_acc[dt] = wmma_f16(Am, cat16(blo, bhi), o_acc[dt]);
            }
        }
        __syncthreads();
    }

    // store o tile: wave l owns cols [l*64, l*64+64)
#pragma unroll
    for (int dt = 0; dt < 4; ++dt) {
#pragma unroll
        for (int r = 0; r < 8; ++r) {
            int row = bb * CN + n0 + r + mAdd;
            o16[(size_t)row * (CL * CD) + l * CD + dt * 16 + lhalf] =
                (_Float16)o_acc[dt][r];
        }
    }
}

// ---------------------------------------------------------------------------
// Kernel 3: out = o_f16 (8192 x 1024) @ WoutT (512 x 1024) -> f32 (8192 x 512)
// ---------------------------------------------------------------------------
__global__ __launch_bounds__(256) void out_gemm(
    const _Float16* __restrict__ o16,
    const _Float16* __restrict__ woutT,   // (512 x 1024) [col][k]
    float* __restrict__ out)
{
    __shared__ __align__(16) _Float16 lds_a[16 * 1024];   // 32KB
    const int t    = threadIdx.x;
    const int row0 = blockIdx.x * 16;
    {
        const _Float16* src = o16 + (size_t)row0 * 1024;
#pragma unroll
        for (int i = 0; i < 8; ++i) {
            int idx = (t + i * 256) * 8;
            *(v8h*)&lds_a[idx] = *(const v8h*)&src[idx];
        }
    }
    __syncthreads();

    const int wave  = t >> 5;
    const int lane  = t & 31;
    const int lhalf = lane & 15;
    const int koff  = (lane < 16) ? 0 : 8;
    const int col0  = (blockIdx.y * 8 + wave) * 16;     // 0..511
    const _Float16* brow = woutT + (size_t)(col0 + lhalf) * 1024;

    v8f acc = {0.f, 0.f, 0.f, 0.f, 0.f, 0.f, 0.f, 0.f};
#pragma unroll 4
    for (int k0 = 0; k0 < 1024; k0 += 32) {
        v8h alo = *(const v8h*)&lds_a[lhalf * 1024 + k0 + koff];
        v8h ahi = *(const v8h*)&lds_a[lhalf * 1024 + k0 + koff + 16];
        v8h blo = *(const v8h*)&brow[k0 + koff];
        v8h bhi = *(const v8h*)&brow[k0 + koff + 16];
        acc = wmma_f16(cat16(alo, ahi), cat16(blo, bhi), acc);
    }
    const int mAdd = (lane < 16) ? 0 : 8;
#pragma unroll
    for (int r = 0; r < 8; ++r)
        out[(size_t)(row0 + r + mAdd) * CH + col0 + lhalf] = acc[r];
}

// ---------------------------------------------------------------------------
// Host launcher
// ---------------------------------------------------------------------------
extern "C" void kernel_launch(void* const* d_in, const int* in_sizes, int n_in,
                              void* d_out, int out_size, void* d_ws, size_t ws_size,
                              hipStream_t stream) {
    (void)in_sizes; (void)n_in; (void)out_size; (void)ws_size;

    const float* x     = (const float*)d_in[0];
    const float* prior = (const float*)d_in[1];
    const float* eps   = (const float*)d_in[2];
    const float* Wq    = (const float*)d_in[3];
    const float* Wk    = (const float*)d_in[4];
    const float* Wv    = (const float*)d_in[5];
    const float* bv    = (const float*)d_in[6];
    const float* sigma = (const float*)d_in[7];
    const float* Wp1   = (const float*)d_in[8];
    const float* bp1   = (const float*)d_in[9];
    const float* Wp2   = (const float*)d_in[10];
    const float* bp2   = (const float*)d_in[11];
    const float* Wout  = (const float*)d_in[12];
    float* out = (float*)d_out;

    // workspace layout (halves); total ~62 MB
    _Float16* w = (_Float16*)d_ws;
    _Float16* x16   = w;                  // 8192*512
    _Float16* wqT   = x16   + 4194304;    // 512*512
    _Float16* wkT   = wqT   + 262144;     // 512*512
    _Float16* wvT   = wkT   + 262144;     // 1024*512
    _Float16* woutT = wvT   + 524288;     // 512*1024
    _Float16* q16   = woutT + 524288;     // 8192*512
    _Float16* k16   = q16   + 4194304;    // 8192*512
    _Float16* vT16  = k16   + 4194304;    // 16*16*64*512
    _Float16* o16   = vT16  + 8388608;    // 8192*1024

    // conversions
    cvt_copy_f16<<<dim3(4194304 / 256), dim3(256), 0, stream>>>(x, x16, 4194304);
    cvt_transpose_f16<<<dim3(262144 / 256), dim3(256), 0, stream>>>(Wq, wqT, 512, 512);
    cvt_transpose_f16<<<dim3(262144 / 256), dim3(256), 0, stream>>>(Wk, wkT, 512, 512);
    cvt_transpose_f16<<<dim3(524288 / 256), dim3(256), 0, stream>>>(Wv, wvT, 512, 1024);
    cvt_transpose_f16<<<dim3(524288 / 256), dim3(256), 0, stream>>>(Wout, woutT, 1024, 512);

    // q/k/v projections (2048 virtual columns)
    qkv_gemm<<<dim3(CM_TOT / 16, 16), dim3(256), 0, stream>>>(
        x16, wqT, wkT, wvT, bv, q16, k16, vT16);

    // fused attention
    attn_fused<<<dim3(CB * (CN / 16)), dim3(512), 0, stream>>>(
        q16, k16, vT16, eps, prior, sigma, Wp1, bp1, Wp2, bp2, o16);

    // output projection
    out_gemm<<<dim3(CM_TOT / 16, 4), dim3(256), 0, stream>>>(o16, woutT, out);
}
